// LowRankHighOrderSelfAttention_67061619360469
// MI455X (gfx1250) — matrix operand order repaired and checked by reading, not compiled, not run on hardware
//
#include <hip/hip_runtime.h>
#include <hip/hip_bf16.h>

// ---------------------------------------------------------------------------
// LowRankHighOrderSelfAttention for MI455X (gfx1250, wave32, WMMA + async LDS)
// B=4, N1=64, N2=128, D=256, NH=8, DH=32, M=192
// ---------------------------------------------------------------------------

typedef __attribute__((ext_vector_type(16))) _Float16 v16h;
typedef __attribute__((ext_vector_type(8)))  float    v8f;
typedef __attribute__((ext_vector_type(4)))  int      v4i;

static constexpr int Bc  = 4;
static constexpr int N1c = 64;
static constexpr int N2c = 128;
static constexpr int Dc  = 256;
static constexpr int NHc = 8;
static constexpr int DHc = 32;
static constexpr int Mc  = 192;
static constexpr int ROWS = Bc * N1c * N2c;           // 32768
static constexpr size_t PLANE = (size_t)ROWS * Dc;    // 8388608 elements

// ---------------------------------------------------------------------------
// CDNA5 async global->LDS copy (16B per lane), with safe fallback.
// Probed signature: (v4i addrspace(1)*, v4i addrspace(3)*, imm offset, imm cpol)
// ---------------------------------------------------------------------------
typedef __attribute__((address_space(1))) v4i gv4i;
typedef __attribute__((address_space(3))) v4i lv4i;

__device__ inline void async_copy_b128(void* lds_dst, const void* gsrc) {
#if __has_builtin(__builtin_amdgcn_global_load_async_to_lds_b128)
  void* g = (void*)(size_t)gsrc;
  __builtin_amdgcn_global_load_async_to_lds_b128((gv4i*)g, (lv4i*)lds_dst, 0, 0);
#else
  *(float4*)lds_dst = *(const float4*)gsrc;
#endif
}

__device__ inline void async_wait_all() {
#if __has_builtin(__builtin_amdgcn_s_wait_asynccnt)
  __builtin_amdgcn_s_wait_asynccnt(0);
#else
  asm volatile("s_wait_asynccnt 0" ::: "memory");
#endif
}

// ---------------------------------------------------------------------------
// f32 -> f16 conversion pass
// ---------------------------------------------------------------------------
__global__ void cvt_f16_kernel(const float* __restrict__ src,
                               _Float16* __restrict__ dst, int n) {
  int i = blockIdx.x * blockDim.x + threadIdx.x;
  if (i < n) dst[i] = (_Float16)src[i];
}

// ---------------------------------------------------------------------------
// NT GEMM, f16 operands, f32 accumulate:
//   C[rows x N] = A[rows x K] * W[N x K]^T + bias
// Each wave computes a 16x64 strip (4 WMMA accumulators, A fragment reused 4x).
// Block = 256 (8 waves) covers 32 rows x 256 cols. grid = rows/32. N must be 256.
// ---------------------------------------------------------------------------
__global__ void gemm16_kernel(const _Float16* __restrict__ A,
                              const _Float16* __restrict__ W,
                              const float* __restrict__ bias,
                              float* __restrict__ C, int K, int N) {
  const int wave = threadIdx.x >> 5;
  const int lane = threadIdx.x & 31;
  const int ln = lane & 15;      // row-within-tile (A) / col-within-tile (B,C)
  const int kh = lane >> 4;      // K-half selector
  const int m0 = blockIdx.x * 32 + (wave >> 2) * 16;
  const int n0 = (wave & 3) * 64;

  const _Float16* ar = A + (size_t)(m0 + ln) * K;
  const _Float16* br = W + (size_t)(n0 + ln) * K;   // strip base, rows n0..n0+63

  v8f acc0 = {}, acc1 = {}, acc2 = {}, acc3 = {};
  for (int k0 = 0; k0 < K; k0 += 32) {
    __builtin_prefetch(ar + k0 + 64, 0, 1);
    v16h a, b0, b1, b2, b3;
#pragma unroll
    for (int x = 0; x < 8; ++x) {
      a[x]     = ar[k0 + kh * 8 + x];          // K = kh*8 + x
      a[8 + x] = ar[k0 + 16 + kh * 8 + x];     // K = 16 + kh*8 + x
    }
#pragma unroll
    for (int x = 0; x < 16; ++x) {
      const int kk = k0 + (x & 8) * 2 + kh * 8 + (x & 7);  // K = table order
      b0[x] = br[kk];
      b1[x] = br[(size_t)16 * K + kk];
      b2[x] = br[(size_t)32 * K + kk];
      b3[x] = br[(size_t)48 * K + kk];
    }
    acc0 = __builtin_amdgcn_wmma_f32_16x16x32_f16(false, a, false, b0, (short)0, acc0, false, false);
    acc1 = __builtin_amdgcn_wmma_f32_16x16x32_f16(false, a, false, b1, (short)0, acc1, false, false);
    acc2 = __builtin_amdgcn_wmma_f32_16x16x32_f16(false, a, false, b2, (short)0, acc2, false, false);
    acc3 = __builtin_amdgcn_wmma_f32_16x16x32_f16(false, a, false, b3, (short)0, acc3, false, false);
  }
  float* crow = C + (size_t)(m0 + kh * 8) * N;
#pragma unroll
  for (int i = 0; i < 8; ++i) {
    float* cr = crow + (size_t)i * N;
    const int c0 = n0 + ln;
    cr[c0]      = acc0[i] + bias[c0];
    cr[c0 + 16] = acc1[i] + bias[c0 + 16];
    cr[c0 + 32] = acc2[i] + bias[c0 + 32];
    cr[c0 + 48] = acc3[i] + bias[c0 + 48];
  }
}

// ---------------------------------------------------------------------------
// Pooling: sum over n2 -> out [B][N1][256]   (for dim=1 attention)
// ---------------------------------------------------------------------------
__global__ void pool_n2_kernel(const float* __restrict__ src, float* __restrict__ dst) {
  int o = blockIdx.x * blockDim.x + threadIdx.x;
  if (o >= Bc * N1c * Dc) return;
  int c = o % Dc;
  int bn1 = o / Dc;
  const float* p = src + (size_t)bn1 * N2c * Dc + c;
  float s = 0.f;
  for (int j = 0; j < N2c; ++j) s += p[(size_t)j * Dc];
  dst[o] = s;
}

// Pooling: sum over n1 -> out [B][N2][256]   (for dim=2 attention)
__global__ void pool_n1_kernel(const float* __restrict__ src, float* __restrict__ dst) {
  int o = blockIdx.x * blockDim.x + threadIdx.x;
  if (o >= Bc * N2c * Dc) return;
  int c = o % Dc;
  int n2 = (o / Dc) % N2c;
  int b = o / (Dc * N2c);
  const float* p = src + (size_t)b * N1c * N2c * Dc + (size_t)n2 * Dc + c;
  float s = 0.f;
  for (int j = 0; j < N1c; ++j) s += p[(size_t)j * N2c * Dc];
  dst[o] = s;
}

// ---------------------------------------------------------------------------
// RoPE, in place on pooled tensor x [B][L][NH][DH]
// ---------------------------------------------------------------------------
__global__ void rope_kernel(float* __restrict__ x, int L) {
  int idx = blockIdx.x * blockDim.x + threadIdx.x;
  int total = Bc * L * NHc * 16;
  if (idx >= total) return;
  int i = idx & 15;          // frequency index
  int rest = idx >> 4;       // (b*L + l)*NH + h
  int l = (rest >> 3) % L;
  float invf = exp2f(-(float)i * (13.287712379549449f / 16.0f)); // 10000^(-i/16)
  float ang = (float)l * invf;
  float c = __cosf(ang), s = __sinf(ang);
  size_t base = (size_t)rest * DHc;
  float x0 = x[base + i], x1 = x[base + 16 + i];
  x[base + i]      = x0 * c - x1 * s;
  x[base + 16 + i] = x1 * c + x0 * s;
}

// ---------------------------------------------------------------------------
// Global min/max of n floats -> out[0]=min, out[1]=max.  Single block.
// ---------------------------------------------------------------------------
__global__ void minmax_kernel(const float* __restrict__ x, int n, float* __restrict__ out) {
  __shared__ float smn[256], smx[256];
  float mn = 3.4e38f, mx = -3.4e38f;
  for (int i = threadIdx.x; i < n; i += 256) {
    float v = x[i];
    mn = fminf(mn, v);
    mx = fmaxf(mx, v);
  }
  smn[threadIdx.x] = mn; smx[threadIdx.x] = mx;
  __syncthreads();
  for (int s = 128; s > 0; s >>= 1) {
    if (threadIdx.x < (unsigned)s) {
      smn[threadIdx.x] = fminf(smn[threadIdx.x], smn[threadIdx.x + s]);
      smx[threadIdx.x] = fmaxf(smx[threadIdx.x], smx[threadIdx.x + s]);
    }
    __syncthreads();
  }
  if (threadIdx.x == 0) { out[0] = smn[0]; out[1] = smx[0]; }
}

// ---------------------------------------------------------------------------
// Performer softmax-kernel features.
//   x   : pooled [B][L][NH][DH]
//   out : f16, base block per (b,h) of L*M; element at l*sL + m*sM
// grid = (L, NH, B), block = 192 (= M)
// ---------------------------------------------------------------------------
__global__ void feat_kernel(const float* __restrict__ x,
                            const float* __restrict__ mm,
                            const float* __restrict__ feat,
                            _Float16* __restrict__ out,
                            int L, int sL, int sM) {
  __shared__ float xs[DHc];
  const int l = blockIdx.x, h = blockIdx.y, b = blockIdx.z;
  const int t = threadIdx.x;                       // m index, 0..191
  const float mn = mm[0], mx = mm[1];
  const float scale = 1.0f / (mx - mn);
  const float invTemp = 0.4204482076268573f;       // 32^(-1/4)
  if (t < DHc) {
    float v = x[(((size_t)b * L + l) * NHc + h) * DHc + t];
    xs[t] = (v - mn) * scale * invTemp;
  }
  __syncthreads();
  float ss = 0.f;
#pragma unroll
  for (int d = 0; d < DHc; ++d) ss += xs[d] * xs[d];
  float off = -0.5f * ss - 0.5f * 5.257495372027781f + 1e-6f;  // ln(192)
  float dot = 0.f;
  const float* fr = feat + (size_t)t * DHc;
#pragma unroll
  for (int d = 0; d < DHc; ++d) dot += xs[d] * fr[d];
  float val = __expf(dot + off);
  out[((size_t)(b * NHc + h)) * L * Mc + (size_t)l * sL + (size_t)t * sM] = (_Float16)val;
}

// kpsum[b][h][m] = sum_l kpT[b][h][m][l]
__global__ void kpsum_kernel(const _Float16* __restrict__ kpT,
                             float* __restrict__ kpsum, int L) {
  int idx = blockIdx.x * blockDim.x + threadIdx.x;   // (b*NH+h)*M + m
  if (idx >= Bc * NHc * Mc) return;
  const _Float16* p = kpT + (size_t)idx * L;
  float s = 0.f;
  for (int l = 0; l < L; ++l) s += (float)p[l];
  kpsum[idx] = s;
}

// rnorm[b][h][l] = 1 / sum_m qp[b][h][l][m] * kpsum[b][h][m]
__global__ void rnorm_kernel(const _Float16* __restrict__ qp,
                             const float* __restrict__ kpsum,
                             float* __restrict__ rnorm, int L) {
  int idx = blockIdx.x * blockDim.x + threadIdx.x;   // (b*NH+h)*L + l
  if (idx >= Bc * NHc * L) return;
  int bh = idx / L;
  const _Float16* q = qp + (size_t)idx * Mc;
  const float* ks = kpsum + (size_t)bh * Mc;
  float s = 0.f;
  for (int m = 0; m < Mc; ++m) s += (float)q[m] * ks[m];
  rnorm[idx] = 1.0f / s;
}

// ---------------------------------------------------------------------------
// Fused linear attention for one (b, j, h):
//   stage 0: async-DMA the V slice [L x 32] (f32) into LDS
//   stage 1: kv[192x32] = kpT[192xL] @ V[Lx32]        (WMMA -> LDS f16)
//   stage 2: out[Lx32]  = (qp[Lx192] @ kv[192x32]) * rnorm[l]
// grid = (Nj, NH, B), block = 256 (8 waves)
// ---------------------------------------------------------------------------
__global__ void attn_kernel(const _Float16* __restrict__ qp,    // [B][NH][L][M]
                            const _Float16* __restrict__ kpT,   // [B][NH][M][L]
                            const float* __restrict__ Vin,      // [B][N1][N2][NH][DH]
                            const float* __restrict__ rnorm,    // [B][NH][L]
                            float* __restrict__ Vout,
                            int L, int vStrideL, int vStrideJ) {
  __shared__ float    vLDS[N2c * DHc];               // 16 KB (max L = 128)
  __shared__ _Float16 kvLDS[Mc * DHc];               // 12 KB
  const int j = blockIdx.x, h = blockIdx.y, b = blockIdx.z;
  const int wave = threadIdx.x >> 5;
  const int lane = threadIdx.x & 31;
  const int ln = lane & 15;
  const int kh = lane >> 4;

  const size_t bStride = (size_t)N1c * N2c * NHc * DHc;
  const float* vBase = Vin + (size_t)b * bStride + (size_t)j * vStrideJ + h * DHc;

  // ---- stage 0: DMA the V slice into LDS (coalesced 16B chunks) ----
  const int chunks = L * (DHc / 4);                  // 16B chunks
  for (int c = threadIdx.x; c < chunks; c += 256) {
    const int l  = c >> 3;
    const int q4 = (c & 7) * 4;
    async_copy_b128(&vLDS[l * DHc + q4], vBase + (size_t)l * vStrideL + q4);
  }
  async_wait_all();
  __syncthreads();

  // ---- stage 1: kv = kpT @ V  (24 tiles: 12 along m, 2 along d) ----
  const _Float16* kpBase = kpT + (size_t)(b * NHc + h) * Mc * L;
  for (int t = wave; t < 24; t += 8) {
    const int m0 = (t % 12) * 16;
    const int n0 = (t / 12) * 16;
    v8f acc = {};
    for (int k0 = 0; k0 < L; k0 += 32) {
      v16h a, bfrag;
      const _Float16* ar = kpBase + (size_t)(m0 + ln) * L + k0;
#pragma unroll
      for (int x = 0; x < 8; ++x) {
        a[x]     = ar[kh * 8 + x];
        a[8 + x] = ar[16 + kh * 8 + x];
      }
      const int n = n0 + ln;
#pragma unroll
      for (int x = 0; x < 16; ++x)
        bfrag[x] = (_Float16)vLDS[(k0 + (x & 8) * 2 + kh * 8 + (x & 7)) * DHc + n];
      acc = __builtin_amdgcn_wmma_f32_16x16x32_f16(false, a, false, bfrag,
                                                   (short)0, acc, false, false);
    }
#pragma unroll
    for (int i = 0; i < 8; ++i)
      kvLDS[(m0 + kh * 8 + i) * DHc + (n0 + ln)] = (_Float16)acc[i];
  }
  __syncthreads();

  // ---- stage 2: att = qp @ kv, scale, store ----
  const _Float16* qpBase = qp + (size_t)(b * NHc + h) * L * Mc;
  const float* rn = rnorm + (size_t)(b * NHc + h) * L;
  float* oBase = Vout + (size_t)b * bStride + (size_t)j * vStrideJ + h * DHc;
  const int lt = L / 16;
  const int nTiles = lt * 2;
  for (int t = wave; t < nTiles; t += 8) {
    const int l0 = (t % lt) * 16;
    const int n0 = (t / lt) * 16;
    v8f acc = {};
    for (int k0 = 0; k0 < Mc; k0 += 32) {
      v16h a, bfrag;
      const _Float16* ar = qpBase + (size_t)(l0 + ln) * Mc + k0;
#pragma unroll
      for (int x = 0; x < 8; ++x) {
        a[x]     = ar[kh * 8 + x];
        a[8 + x] = ar[16 + kh * 8 + x];
      }
      const int n = n0 + ln;
#pragma unroll
      for (int x = 0; x < 16; ++x)
        bfrag[x] = kvLDS[(k0 + (x & 8) * 2 + kh * 8 + (x & 7)) * DHc + n];
      acc = __builtin_amdgcn_wmma_f32_16x16x32_f16(false, a, false, bfrag,
                                                   (short)0, acc, false, false);
    }
#pragma unroll
    for (int i = 0; i < 8; ++i) {
      const int l = l0 + kh * 8 + i;
      oBase[(size_t)l * vStrideL + (n0 + ln)] = acc[i] * rn[l];
    }
  }
}

// ---------------------------------------------------------------------------
// Host launcher
// ---------------------------------------------------------------------------
extern "C" void kernel_launch(void* const* d_in, const int* in_sizes, int n_in,
                              void* d_out, int out_size, void* d_ws, size_t ws_size,
                              hipStream_t stream) {
  const float* X    = (const float*)d_in[0];
  const float* Wq   = (const float*)d_in[1];
  const float* bq   = (const float*)d_in[2];
  const float* Wk   = (const float*)d_in[3];
  const float* bk   = (const float*)d_in[4];
  const float* Wv   = (const float*)d_in[5];
  const float* bv   = (const float*)d_in[6];
  const float* Wo   = (const float*)d_in[7];
  const float* bo   = (const float*)d_in[8];
  const float* feat = (const float*)d_in[9];
  float* out = (float*)d_out;

  // workspace layout
  float* w = (float*)d_ws;
  float* stage = w;                       // Q proj, then K proj, then V2
  float* Vv    = w + PLANE;               // value projection
  float* V1    = w + 2 * PLANE;           // output of dim=1 attention
  float* q1p   = w + 3 * PLANE;           // [B][64][8][32]
  float* k1p   = q1p + Bc * N1c * Dc;
  float* q2p   = k1p + Bc * N1c * Dc;     // [B][128][8][32]
  float* k2p   = q2p + Bc * N2c * Dc;
  float* ksum1 = k2p + Bc * N2c * Dc;     // [B][8][192]
  float* ksum2 = ksum1 + Bc * NHc * Mc;
  float* rn1   = ksum2 + Bc * NHc * Mc;   // [B][8][64]
  float* rn2   = rn1 + Bc * NHc * N1c;    // [B][8][128]
  float* mm    = rn2 + Bc * NHc * N2c;    // 4 (min,max) pairs
  _Float16* qp1  = (_Float16*)(mm + 16);
  _Float16* kpT1 = qp1 + (size_t)Bc * NHc * N1c * Mc;
  _Float16* qp2  = kpT1 + (size_t)Bc * NHc * N1c * Mc;
  _Float16* kpT2 = qp2 + (size_t)Bc * NHc * N2c * Mc;
  _Float16* Xh   = kpT2 + (size_t)Bc * NHc * N2c * Mc; // f16 X; reused as f16 V2
  _Float16* Wqh  = Xh + PLANE;
  _Float16* Wkh  = Wqh + Dc * Dc;
  _Float16* Wvh  = Wkh + Dc * Dc;
  _Float16* Woh  = Wvh + Dc * Dc;

  const int NW = Dc * Dc;                 // 65536 weight elements

  // f16 conversions (X once; weights once)
  cvt_f16_kernel<<<(int)((PLANE + 255) / 256), 256, 0, stream>>>(X, Xh, (int)PLANE);
  cvt_f16_kernel<<<(NW + 255) / 256, 256, 0, stream>>>(Wq, Wqh, NW);
  cvt_f16_kernel<<<(NW + 255) / 256, 256, 0, stream>>>(Wk, Wkh, NW);
  cvt_f16_kernel<<<(NW + 255) / 256, 256, 0, stream>>>(Wv, Wvh, NW);
  cvt_f16_kernel<<<(NW + 255) / 256, 256, 0, stream>>>(Wo, Woh, NW);

  const dim3 gGemm(ROWS / 32);

  // Q projection -> stage; pool both ways
  gemm16_kernel<<<gGemm, 256, 0, stream>>>(Xh, Wqh, bq, stage, Dc, Dc);
  pool_n2_kernel<<<(Bc * N1c * Dc + 255) / 256, 256, 0, stream>>>(stage, q1p);
  pool_n1_kernel<<<(Bc * N2c * Dc + 255) / 256, 256, 0, stream>>>(stage, q2p);

  // K projection -> stage (reuse); pool both ways
  gemm16_kernel<<<gGemm, 256, 0, stream>>>(Xh, Wkh, bk, stage, Dc, Dc);
  pool_n2_kernel<<<(Bc * N1c * Dc + 255) / 256, 256, 0, stream>>>(stage, k1p);
  pool_n1_kernel<<<(Bc * N2c * Dc + 255) / 256, 256, 0, stream>>>(stage, k2p);

  // V projection
  gemm16_kernel<<<gGemm, 256, 0, stream>>>(Xh, Wvh, bv, Vv, Dc, Dc);

  // RoPE on dim=2 pooled q/k
  const int ropeN = Bc * N2c * NHc * 16;
  rope_kernel<<<(ropeN + 255) / 256, 256, 0, stream>>>(q2p, N2c);
  rope_kernel<<<(ropeN + 255) / 256, 256, 0, stream>>>(k2p, N2c);

  // global min/max
  minmax_kernel<<<1, 256, 0, stream>>>(q1p, Bc * N1c * Dc, mm + 0);
  minmax_kernel<<<1, 256, 0, stream>>>(k1p, Bc * N1c * Dc, mm + 2);
  minmax_kernel<<<1, 256, 0, stream>>>(q2p, Bc * N2c * Dc, mm + 4);
  minmax_kernel<<<1, 256, 0, stream>>>(k2p, Bc * N2c * Dc, mm + 6);

  // feature maps (q row-major [l][m], k transposed [m][l])
  feat_kernel<<<dim3(N1c, NHc, Bc), Mc, 0, stream>>>(q1p, mm + 0, feat, qp1,  N1c, Mc, 1);
  feat_kernel<<<dim3(N1c, NHc, Bc), Mc, 0, stream>>>(k1p, mm + 2, feat, kpT1, N1c, 1, N1c);
  feat_kernel<<<dim3(N2c, NHc, Bc), Mc, 0, stream>>>(q2p, mm + 4, feat, qp2,  N2c, Mc, 1);
  feat_kernel<<<dim3(N2c, NHc, Bc), Mc, 0, stream>>>(k2p, mm + 6, feat, kpT2, N2c, 1, N2c);

  // key feature sums and reciprocal normalizers
  kpsum_kernel<<<(Bc * NHc * Mc + 255) / 256, 256, 0, stream>>>(kpT1, ksum1, N1c);
  kpsum_kernel<<<(Bc * NHc * Mc + 255) / 256, 256, 0, stream>>>(kpT2, ksum2, N2c);
  rnorm_kernel<<<(Bc * NHc * N1c + 255) / 256, 256, 0, stream>>>(qp1, ksum1, rn1, N1c);
  rnorm_kernel<<<(Bc * NHc * N2c + 255) / 256, 256, 0, stream>>>(qp2, ksum2, rn2, N2c);

  // dim=1 attention: l = n1 (stride 32768), j = n2 (stride 256)
  attn_kernel<<<dim3(N2c, NHc, Bc), 256, 0, stream>>>(
      qp1, kpT1, Vv, rn1, V1, N1c, N2c * NHc * DHc, NHc * DHc);
  // dim=2 attention: l = n2 (stride 256), j = n1 (stride 32768); V2 -> stage
  attn_kernel<<<dim3(N1c, NHc, Bc), 256, 0, stream>>>(
      qp2, kpT2, V1, rn2, stage, N2c, NHc * DHc, N2c * NHc * DHc);

  // output projection: convert V2 to f16 (reuse Xh) then GEMM
  cvt_f16_kernel<<<(int)((PLANE + 255) / 256), 256, 0, stream>>>(stage, Xh, (int)PLANE);
  gemm16_kernel<<<gGemm, 256, 0, stream>>>(Xh, Woh, bo, out, Dc, Dc);

  (void)in_sizes; (void)n_in; (void)out_size; (void)ws_size;
}
